// Bidirectional_GRU_Decoder_19078244729392
// MI455X (gfx1250) — compile-verified, matrix operand order checked by reading, not run on hardware
//
#include <hip/hip_runtime.h>
#include <math.h>

// Problem constants (B, S, D, H, L) = (64, 512, 512, 512, 2)
#define BB 64
#define SS 512
#define DD 512
#define HH 512
#define G3H 1536            // 3*H
#define WMAT (1536 * 512)   // elements per weight matrix
#define BHE (BB * HH)       // 32768 elements per hidden state
#define GBUF (BB * G3H)     // 98304 elements per GEMM output
#define NBLK 48
#define NTHR 256
#define NTID (NBLK * NTHR)  // 12288 threads in the persistent grid

typedef __bf16 bf16_t;
typedef __attribute__((ext_vector_type(16))) __bf16 v16bf;
typedef __attribute__((ext_vector_type(8)))  __bf16 v8bf;
typedef __attribute__((ext_vector_type(8)))  float  v8f;

// ---------------------------------------------------------------------------
// WMMA fragment helpers (wave32, CDNA5 16x16x32 bf16 layout per CDNA5 ISA).
// A fragment (16x32, M x K): lane L holds row (L&15); lanes 0-15 carry
// K = k..k+7 / k+16..k+23, lanes 16-31 carry K = k+8..k+15 / k+24..k+31.
// B fragment (32x16, K x N) for act @ W.T: lane L holds column n = (L&15),
// i.e. row n of W read contiguously along k with the same +8 lane split.
// Both therefore share one loader over a row-major bf16 matrix.
// ---------------------------------------------------------------------------
__device__ __forceinline__ v16bf load_frag(const bf16_t* base, int ld,
                                           int row_base, int k_base) {
  const int lane = threadIdx.x & 31;
  const int r  = row_base + (lane & 15);
  const int ko = k_base + ((lane & 16) >> 1);   // +8 for upper half-wave
  const bf16_t* p = base + (size_t)r * ld + ko;
  v8bf lo = *(const v8bf*)(p);        // 16B -> global_load_b128
  v8bf hi = *(const v8bf*)(p + 16);   // 16B -> global_load_b128
  return __builtin_shufflevector(lo, hi, 0, 1, 2, 3, 4, 5, 6, 7,
                                         8, 9, 10, 11, 12, 13, 14, 15);
}

// C/D fragment (16x16 f32): lane L holds column (L&15); VGPR v holds row
// v + (L>=16 ? 8 : 0).
__device__ __forceinline__ void store_tile(float* C, int ldc, int row_base,
                                           int col_base, v8f acc) {
  const int lane = threadIdx.x & 31;
  const int n  = col_base + (lane & 15);
  const int r0 = row_base + ((lane & 16) >> 1);
#pragma unroll
  for (int v = 0; v < 8; ++v) C[(size_t)(r0 + v) * ldc + n] = acc[v];
}

#define WMMA_BF16(A, Bf, Cf) \
  __builtin_amdgcn_wmma_f32_16x16x32_bf16(false, (A), false, (Bf), (short)0, (Cf), false, false)

// ---------------------------------------------------------------------------
// Grid-wide split barrier: monotonic atomic counter in d_ws. All NBLK blocks
// are co-resident (12K threads on a many-WGP part), so spinning is safe.
// Release (fence + add) / acquire (spin + fence) makes G / hbf stores from
// other WGPs visible through L2.
// ---------------------------------------------------------------------------
__device__ __forceinline__ void grid_barrier(unsigned* cnt, unsigned target) {
  __syncthreads();
  if (threadIdx.x == 0) {
    __threadfence();                                  // release our stores
    __atomic_fetch_add(cnt, 1u, __ATOMIC_RELAXED);
    while (__atomic_load_n(cnt, __ATOMIC_RELAXED) < target)
      __builtin_amdgcn_s_sleep(1);
    __threadfence();                                  // acquire their stores
  }
  __syncthreads();
}

// ---------------------------------------------------------------------------
// Per-timestep fused kernel arguments.
// Wbf layout: [0]=Wihf L0 [1]=Wihf L1 [2]=Whhf L0 [3]=Whhf L1
//             [4]=Wihb L0 (k-reversed) [5]=Wihb L1 [6]=Whhb L0 [7]=Whhb L1
// h32/hbf layout: [h0f][h1f][h0b][h1b], each B*H  -> offset (layer + 2*dir)*BHE
// G layout: [gi0f][gh0f][gi0b][gh0b][gi1f][gh1f][gi1b][gh1b], each B*3H
// ---------------------------------------------------------------------------
struct StepArgs {
  const bf16_t* Xbf;
  const bf16_t* Wbf;
  float*  h32;
  bf16_t* hbf;
  float*  G;
  const float* bihf; const float* bhhf;
  const float* bihb; const float* bhhb;
  float* out;
  unsigned* barrier;
  int t;
};

// One 16x64 output strip per wave; K-loop register double-buffered so the
// k+32 fragment loads issue before the k WMMAs (hides L2 latency).
__device__ __forceinline__ void gemm_phase(const StepArgs& s, int layer) {
  const int w = blockIdx.x * 8 + (threadIdx.x >> 5);  // 0..383
  const int which  = w / 96;                          // which of 4 GEMMs
  const int wl     = w % 96;
  const int m_base = (wl / 24) * 16;                  // M = 64 -> 4 tiles
  const int n_base = (wl % 24) * 64;                  // N = 1536 -> 24 strips

  const bf16_t* A; int lda; const bf16_t* W; float* C;
  if (layer == 0) {
    switch (which) {
      case 0:  A = s.Xbf + (size_t)s.t * DD; lda = SS * DD;
               W = s.Wbf + 0 * (size_t)WMAT; break;   // gi0_f
      case 1:  A = s.hbf + 0 * BHE; lda = HH;
               W = s.Wbf + 2 * (size_t)WMAT; break;   // gh0_f (h0f)
      case 2:  A = s.Xbf + (size_t)s.t * DD; lda = SS * DD;
               W = s.Wbf + 4 * (size_t)WMAT; break;   // gi0_b (rev W)
      default: A = s.hbf + 2 * BHE; lda = HH;
               W = s.Wbf + 6 * (size_t)WMAT; break;   // gh0_b (h0b)
    }
    C = s.G + (size_t)which * GBUF;
  } else {
    switch (which) {
      case 0:  A = s.hbf + 0 * BHE; W = s.Wbf + 1 * (size_t)WMAT; break; // gi1_f
      case 1:  A = s.hbf + 1 * BHE; W = s.Wbf + 3 * (size_t)WMAT; break; // gh1_f
      case 2:  A = s.hbf + 2 * BHE; W = s.Wbf + 5 * (size_t)WMAT; break; // gi1_b
      default: A = s.hbf + 3 * BHE; W = s.Wbf + 7 * (size_t)WMAT; break; // gh1_b
    }
    lda = HH;
    C = s.G + (size_t)(4 + which) * GBUF;
  }

  v8f c0 = {}, c1 = {}, c2 = {}, c3 = {};
  v16bf a  = load_frag(A, lda, m_base, 0);
  v16bf b0 = load_frag(W, 512, n_base +  0, 0);
  v16bf b1 = load_frag(W, 512, n_base + 16, 0);
  v16bf b2 = load_frag(W, 512, n_base + 32, 0);
  v16bf b3 = load_frag(W, 512, n_base + 48, 0);
#pragma unroll 3
  for (int k = 32; k < 512; k += 32) {
    v16bf a2  = load_frag(A, lda, m_base, k);
    v16bf nb0 = load_frag(W, 512, n_base +  0, k);
    v16bf nb1 = load_frag(W, 512, n_base + 16, k);
    v16bf nb2 = load_frag(W, 512, n_base + 32, k);
    v16bf nb3 = load_frag(W, 512, n_base + 48, k);
    c0 = WMMA_BF16(a, b0, c0);
    c1 = WMMA_BF16(a, b1, c1);
    c2 = WMMA_BF16(a, b2, c2);
    c3 = WMMA_BF16(a, b3, c3);
    a = a2; b0 = nb0; b1 = nb1; b2 = nb2; b3 = nb3;
  }
  c0 = WMMA_BF16(a, b0, c0);
  c1 = WMMA_BF16(a, b1, c1);
  c2 = WMMA_BF16(a, b2, c2);
  c3 = WMMA_BF16(a, b3, c3);

  store_tile(C, G3H, m_base, n_base +  0, c0);
  store_tile(C, G3H, m_base, n_base + 16, c1);
  store_tile(C, G3H, m_base, n_base + 32, c2);
  store_tile(C, G3H, m_base, n_base + 48, c3);
}

__device__ __forceinline__ void gates_phase(const StepArgs& s, int layer) {
  const int gbase = layer ? 4 : 0;
  for (int idx = blockIdx.x * NTHR + threadIdx.x; idx < 2 * BHE; idx += NTID) {
    const int dir = idx >> 15;                 // B*H = 32768
    const int e   = idx & (BHE - 1);
    const int b   = e >> 9;
    const int j   = e & (HH - 1);

    const float* GI  = s.G + (size_t)(gbase + 2 * dir) * GBUF;
    const float* GH  = GI + GBUF;
    const float* bih = (dir ? s.bihb : s.bihf) + layer * G3H;
    const float* bhh = (dir ? s.bhhb : s.bhhf) + layer * G3H;
    float*  h   = s.h32 + (size_t)(layer + 2 * dir) * BHE;
    bf16_t* hbf = s.hbf + (size_t)(layer + 2 * dir) * BHE;

    const size_t g = (size_t)b * G3H;
    const float ir = GI[g + j]        + bih[j];
    const float iz = GI[g + 512 + j]  + bih[512 + j];
    const float in = GI[g + 1024 + j] + bih[1024 + j];
    const float hr = GH[g + j]        + bhh[j];
    const float hz = GH[g + 512 + j]  + bhh[512 + j];
    const float hn = GH[g + 1024 + j] + bhh[1024 + j];

    const float r = 1.0f / (1.0f + __expf(-(ir + hr)));
    const float z = 1.0f / (1.0f + __expf(-(iz + hz)));
    const float n = tanhf(in + r * hn);
    const float hv = (1.0f - z) * n + z * h[e];

    h[e]   = hv;
    hbf[e] = (bf16_t)hv;

    if (layer == 1) {   // emit concat([fwd, bwd]) at timestep t
      const size_t o = (size_t)b * (SS * 2 * HH) + (size_t)s.t * (2 * HH)
                     + (size_t)dir * HH + j;
      s.out[o] = hv;
      if (s.t == SS - 1) {  // second tuple element: out[:, -1, :]
        s.out[(size_t)BB * SS * 2 * HH + (size_t)b * (2 * HH)
              + (size_t)dir * HH + j] = hv;
      }
    }
  }
}

// Fused per-timestep kernel: GEMM-L0 | barrier | gates-L0 | barrier |
// GEMM-L1 | barrier | gates-L1.  Cross-launch ordering (gates-L1 -> next
// step's GEMM-L0) is provided by the stream.
__global__ __launch_bounds__(NTHR) void step_kernel(StepArgs s) {
  const unsigned base = (unsigned)s.t * 3u * NBLK;
  gemm_phase(s, 0);
  grid_barrier(s.barrier, base + 1u * NBLK);
  gates_phase(s, 0);
  grid_barrier(s.barrier, base + 2u * NBLK);
  gemm_phase(s, 1);
  grid_barrier(s.barrier, base + 3u * NBLK);
  gates_phase(s, 1);
}

// ---------------------------------------------------------------------------
// Prep kernels: fp32 -> bf16 conversions, initial state broadcast, barrier
// counter reset (must run every kernel_launch for deterministic replays).
// ---------------------------------------------------------------------------
__global__ __launch_bounds__(256) void convert_input_kernel(const float* x,
                                                            bf16_t* y, int n) {
  const int i = blockIdx.x * 256 + threadIdx.x;
  if (i < n) y[i] = (bf16_t)x[i];
}

__global__ __launch_bounds__(256) void prep_weights_kernel(
    const float* Wihf, const float* Whhf, const float* Wihb, const float* Whhb,
    bf16_t* dst) {
  const long i = (long)blockIdx.x * 256 + threadIdx.x;
  if (i >= 8L * WMAT) return;
  const int m = (int)(i / WMAT);
  const long e = i % WMAT;
  const float* src; long se = e;
  switch (m) {
    case 0: src = Wihf; break;
    case 1: src = Wihf + WMAT; break;
    case 2: src = Whhf; break;
    case 3: src = Whhf + WMAT; break;
    case 4: src = Wihb; se = (e / 512) * 512 + (511 - (e % 512)); break;
    case 5: src = Wihb + WMAT; break;
    case 6: src = Whhb; break;
    default: src = Whhb + WMAT; break;
  }
  dst[i] = (bf16_t)src[se];
}

__global__ __launch_bounds__(256) void init_h_kernel(const float* eh,
                                                     float* h32, bf16_t* hbf,
                                                     unsigned* barrier) {
  const int i = blockIdx.x * 256 + threadIdx.x;
  if (i == 0) *barrier = 0u;     // reset grid-barrier counter every launch
  if (i >= BHE) return;
  const int b = i >> 9, j = i & (HH - 1);
  const float f  = eh[(size_t)b * (2 * HH) + j];
  const float bk = eh[(size_t)b * (2 * HH) + HH + j];
  h32[i] = f;  h32[BHE + i] = f;  h32[2 * BHE + i] = bk;  h32[3 * BHE + i] = bk;
  hbf[i] = (bf16_t)f;            hbf[BHE + i] = (bf16_t)f;
  hbf[2 * BHE + i] = (bf16_t)bk; hbf[3 * BHE + i] = (bf16_t)bk;
}

// ---------------------------------------------------------------------------
// Driver. Workspace (~50 MB):
//   ctr  : barrier counter        (       256 B)
//   Xbf  : B*S*D bf16             (33,554,432 B)
//   Wbf  : 8 * 1536*512 bf16      (12,582,912 B)
//   h32  : 4 * B*H f32            (   524,288 B)
//   hbf  : 4 * B*H bf16           (   262,144 B)
//   G    : 8 * B*3H f32           ( 3,145,728 B)
// ---------------------------------------------------------------------------
extern "C" void kernel_launch(void* const* d_in, const int* in_sizes, int n_in,
                              void* d_out, int out_size, void* d_ws, size_t ws_size,
                              hipStream_t stream) {
  (void)in_sizes; (void)n_in; (void)out_size; (void)ws_size;
  const float* input     = (const float*)d_in[0];
  const float* encoder_h = (const float*)d_in[2];
  const float* W_ih_f = (const float*)d_in[3];
  const float* W_hh_f = (const float*)d_in[4];
  const float* b_ih_f = (const float*)d_in[5];
  const float* b_hh_f = (const float*)d_in[6];
  const float* W_ih_b = (const float*)d_in[7];
  const float* W_hh_b = (const float*)d_in[8];
  const float* b_ih_b = (const float*)d_in[9];
  const float* b_hh_b = (const float*)d_in[10];
  float* out = (float*)d_out;

  char* p = (char*)d_ws;
  unsigned* ctr = (unsigned*)p; p += 256;
  bf16_t* Xbf = (bf16_t*)p; p += (size_t)BB * SS * DD * 2;
  bf16_t* Wbf = (bf16_t*)p; p += (size_t)8 * WMAT * 2;
  float*  h32 = (float*)p;  p += (size_t)4 * BHE * 4;
  bf16_t* hbf = (bf16_t*)p; p += (size_t)4 * BHE * 2;
  float*  G   = (float*)p;

  convert_input_kernel<<<(BB * SS * DD) / 256, 256, 0, stream>>>(
      input, Xbf, BB * SS * DD);
  prep_weights_kernel<<<(8 * WMAT) / 256, 256, 0, stream>>>(
      W_ih_f, W_hh_f, W_ih_b, W_hh_b, Wbf);
  init_h_kernel<<<BHE / 256, 256, 0, stream>>>(encoder_h, h32, hbf, ctr);

  StepArgs s;
  s.Xbf = Xbf; s.Wbf = Wbf; s.h32 = h32; s.hbf = hbf; s.G = G;
  s.bihf = b_ih_f; s.bhhf = b_hh_f; s.bihb = b_ih_b; s.bhhb = b_hh_b;
  s.out = out; s.barrier = ctr;

  for (int t = 0; t < SS; ++t) {
    s.t = t;
    step_kernel<<<NBLK, NTHR, 0, stream>>>(s);
  }
}
